// BertSelfAttention_7962869367489
// MI455X (gfx1250) — compile-verified, matrix operand order checked by reading, not run on hardware
//
#include <hip/hip_runtime.h>
#include <hip/hip_bf16.h>
#include <math.h>

// Problem constants (from reference): B=4, S=1024, K=512, H=1024, NH=16, HD=64
#define NHD   16
#define HD    64
#define HDIM  1024
#define BSZ   4
#define SEQ   1024
#define KSEQ  512

typedef __bf16 bf16_t;
typedef __attribute__((ext_vector_type(16))) __bf16 v16bf;
typedef __attribute__((ext_vector_type(8)))  __bf16 v8bf;
typedef __attribute__((ext_vector_type(8)))  float  v8f;

__device__ inline bf16_t f2bf(float f) {
  union { float f; unsigned int u; } c; c.f = f;
  unsigned int u = c.u + 0x7fffu + ((c.u >> 16) & 1u);  // round-to-nearest-even
  union { unsigned short s; bf16_t b; } o; o.s = (unsigned short)(u >> 16);
  return o.b;
}

// Load a 16x32 bf16 tile in the CDNA5 16-bit A/B WMMA lane layout from a
// row-major [rows, >=32] buffer with row stride `ld` (elements).
// Lane m in [0,16): row m, K {0..7, 16..23}; lane m+16: row m, K {8..15, 24..31}.
__device__ inline v16bf load_tile16x32(const bf16_t* __restrict__ base, int ld) {
  const int lane = threadIdx.x & 31;
  const int r = lane & 15;
  const int k0 = (lane >> 4) << 3;  // 0 or 8
  const bf16_t* p = base + (size_t)r * ld + k0;
  v8bf lo = *(const v8bf*)(p);
  v8bf hi = *(const v8bf*)(p + 16);
  v16bf out;
#pragma unroll
  for (int i = 0; i < 8; ++i) { out[i] = lo[i]; out[i + 8] = hi[i]; }
  return out;
}

__device__ inline v8f wmma_bf16(v16bf a, v16bf b, v8f c) {
  return __builtin_amdgcn_wmma_f32_16x16x32_bf16(false, a, false, b, (short)0, c,
                                                 false, false);
}

// ---------------- conversion kernels ----------------
__global__ void cvt_f32_bf16(const float* __restrict__ in, bf16_t* __restrict__ out,
                             int n) {
  int i = blockIdx.x * blockDim.x + threadIdx.x;
  if (i < n) out[i] = f2bf(in[i]);
}

// Wt[n*H + k] = bf16(W[k*H + n])  (store W transposed for B-matrix tile loads)
__global__ void cvt_transpose_w(const float* __restrict__ W, bf16_t* __restrict__ Wt) {
  int i = blockIdx.x * blockDim.x + threadIdx.x;  // over H*H
  int nrow = i >> 10;
  int kcol = i & 1023;
  Wt[i] = f2bf(W[kcol * HDIM + nrow]);
}

// ---------------- projection GEMM ----------------
// out = X[M,H] * W + bias, W given transposed as Wt[N=H rows, K=H].
// Each wave computes a 32x64 output tile: per K-step, 2 A-tiles + 4 B-tiles
// feed 8 WMMAs (~22 flops/byte from L2). The 64-column group is one head, so
// h == tn4 at compile time. L is a template parameter so the row->(b,l) split
// lowers to shifts instead of the integer-division expansion seen in round 1.
// Output bf16, head-split:
//   TRANSPOSED==0 : [B, NH, L, HD]   (q, k, kq, kk)
//   TRANSPOSED==1 : [B, NH, HD, L]   (v, kv) — ready as PV B-matrix
template <int L, int TRANSPOSED>
__global__ __launch_bounds__(256)
void gemm_proj(const bf16_t* __restrict__ X, const bf16_t* __restrict__ Wt,
               const float* __restrict__ bias, bf16_t* __restrict__ out, int M) {
  const int wave = (int)((blockIdx.x * blockDim.x + threadIdx.x) >> 5);
  const int tilesN4 = HDIM / 64;   // 16 head-sized column groups
  const int tm  = wave / tilesN4;  // 32-row tile
  const int tn4 = wave % tilesN4;  // 64-col tile == head index
  if (tm * 32 >= M) return;
  const int lane = threadIdx.x & 31;
  const bf16_t* A0 = X + (size_t)tm * 32 * HDIM;
  const bf16_t* A1 = A0 + (size_t)16 * HDIM;
  const bf16_t* Bb = Wt + (size_t)tn4 * 64 * HDIM;

  v8f acc[2][4];
#pragma unroll
  for (int mi = 0; mi < 2; ++mi)
#pragma unroll
    for (int ni = 0; ni < 4; ++ni)
      acc[mi][ni] = (v8f){0.f, 0.f, 0.f, 0.f, 0.f, 0.f, 0.f, 0.f};

  for (int k = 0; k < HDIM; k += 32) {
    __builtin_prefetch(A0 + k + 256, 0, 0);  // global_prefetch_b8
    v16bf a0 = load_tile16x32(A0 + k, HDIM);
    v16bf a1 = load_tile16x32(A1 + k, HDIM);
#pragma unroll
    for (int ni = 0; ni < 4; ++ni) {
      v16bf b = load_tile16x32(Bb + (size_t)ni * 16 * HDIM + k, HDIM);
      acc[0][ni] = wmma_bf16(a0, b, acc[0][ni]);
      acc[1][ni] = wmma_bf16(a1, b, acc[1][ni]);
    }
  }

  const int nlo = lane & 15;
  const int hi8 = (lane >> 4) << 3;
  const int h = tn4;
#pragma unroll
  for (int mi = 0; mi < 2; ++mi) {
#pragma unroll
    for (int ni = 0; ni < 4; ++ni) {
      const int d = ni * 16 + nlo;
      const float bv = bias[tn4 * 64 + d];
#pragma unroll
      for (int j = 0; j < 8; ++j) {
        int m = tm * 32 + mi * 16 + j + hi8;  // C layout: row per VGPR slot
        int b_ = m / L, l = m % L;            // L is constexpr -> shift/mask
        float val = acc[mi][ni][j] + bv;
        size_t idx = TRANSPOSED
            ? ((size_t)(b_ * NHD + h) * HD + d) * L + l
            : ((size_t)(b_ * NHD + h) * L + l) * HD + d;
        out[idx] = f2bf(val);
      }
    }
  }
}

// ---------------- flash attention branch (one wave = 16 query rows) ----------------
__device__ void attn_branch(const bf16_t* __restrict__ Qhead, int s0,
                            const bf16_t* __restrict__ Khead,   // [T, HD]
                            const bf16_t* __restrict__ Vthead,  // [HD, T]
                            const float* __restrict__ mask,     // [T]
                            int T, float scale, bf16_t* lds, float* Oout) {
  const int lane = threadIdx.x & 31;
  const int nlo = lane & 15;
  const int hi8 = (lane >> 4) << 3;
  v16bf aq0 = load_tile16x32(Qhead + (size_t)s0 * HD + 0, HD);
  v16bf aq1 = load_tile16x32(Qhead + (size_t)s0 * HD + 32, HD);
  float mrow[8], lrow[8];
#pragma unroll
  for (int j = 0; j < 8; ++j) { mrow[j] = -1.0e30f; lrow[j] = 0.f; }
  v8f O0 = {0.f,0.f,0.f,0.f,0.f,0.f,0.f,0.f};
  v8f O1 = O0, O2 = O0, O3 = O0;

  for (int t = 0; t < T; t += 32) {
    // scores S[16 x 32] = q(16x64) . k^T : 4 WMMAs
    v8f slo = {0.f,0.f,0.f,0.f,0.f,0.f,0.f,0.f};
    v8f shi = slo;
    slo = wmma_bf16(aq0, load_tile16x32(Khead + (size_t)t * HD + 0, HD), slo);
    slo = wmma_bf16(aq1, load_tile16x32(Khead + (size_t)t * HD + 32, HD), slo);
    shi = wmma_bf16(aq0, load_tile16x32(Khead + (size_t)(t + 16) * HD + 0, HD), shi);
    shi = wmma_bf16(aq1, load_tile16x32(Khead + (size_t)(t + 16) * HD + 32, HD), shi);
    const float mklo = mask[t + nlo];
    const float mkhi = mask[t + 16 + nlo];
    float plo[8], phi[8];
#pragma unroll
    for (int j = 0; j < 8; ++j) {
      plo[j] = slo[j] * scale + mklo;
      phi[j] = shi[j] * scale + mkhi;
    }
    // online softmax: row reductions live across 16-lane groups (C layout)
#pragma unroll
    for (int j = 0; j < 8; ++j) {
      float v = fmaxf(plo[j], phi[j]);
      v = fmaxf(v, __shfl_xor(v, 1, 16));
      v = fmaxf(v, __shfl_xor(v, 2, 16));
      v = fmaxf(v, __shfl_xor(v, 4, 16));
      v = fmaxf(v, __shfl_xor(v, 8, 16));
      float mn = fmaxf(mrow[j], v);
      float alpha = __expf(mrow[j] - mn);
      plo[j] = __expf(plo[j] - mn);
      phi[j] = __expf(phi[j] - mn);
      float rs = plo[j] + phi[j];
      rs += __shfl_xor(rs, 1, 16);
      rs += __shfl_xor(rs, 2, 16);
      rs += __shfl_xor(rs, 4, 16);
      rs += __shfl_xor(rs, 8, 16);
      lrow[j] = lrow[j] * alpha + rs;
      mrow[j] = mn;
      O0[j] *= alpha; O1[j] *= alpha; O2[j] *= alpha; O3[j] *= alpha;
    }
    // C-layout -> A-layout transpose of P through LDS (stride 40 keeps rows
    // 16B-aligned and spreads banks)
#pragma unroll
    for (int j = 0; j < 8; ++j) {
      int r = j + hi8;
      lds[r * 40 + nlo] = f2bf(plo[j]);
      lds[r * 40 + nlo + 16] = f2bf(phi[j]);
    }
    asm volatile("s_wait_dscnt 0x0" ::: "memory");  // same-wave cross-lane RAW
    v16bf pa = load_tile16x32(lds, 40);
    // ctx += P(16x32) . V(32x64) : 4 WMMAs (B from v^T, t-contiguous rows)
    O0 = wmma_bf16(pa, load_tile16x32(Vthead + (size_t)0 * T + t, T), O0);
    O1 = wmma_bf16(pa, load_tile16x32(Vthead + (size_t)16 * T + t, T), O1);
    O2 = wmma_bf16(pa, load_tile16x32(Vthead + (size_t)32 * T + t, T), O2);
    O3 = wmma_bf16(pa, load_tile16x32(Vthead + (size_t)48 * T + t, T), O3);
  }
#pragma unroll
  for (int j = 0; j < 8; ++j) {
    float invl = 1.0f / lrow[j];
    Oout[0 * 8 + j] = O0[j] * invl;
    Oout[1 * 8 + j] = O1[j] * invl;
    Oout[2 * 8 + j] = O2[j] * invl;
    Oout[3 * 8 + j] = O3[j] * invl;
  }
}

__global__ __launch_bounds__(256)
void attention_kernel(const bf16_t* __restrict__ qb, const bf16_t* __restrict__ kb,
                      const bf16_t* __restrict__ vtb,
                      const bf16_t* __restrict__ kqb, const bf16_t* __restrict__ kkb,
                      const bf16_t* __restrict__ kvtb,
                      const float* __restrict__ amask, const float* __restrict__ emask,
                      float* __restrict__ out) {
  __shared__ __align__(16) bf16_t lds[8][16 * 40];
  const int wave = (int)((blockIdx.x * blockDim.x + threadIdx.x) >> 5);
  const int wib = threadIdx.x >> 5;
  const int st = wave & 63;               // S/16 tiles
  const int h = (wave >> 6) & (NHD - 1);
  const int b = wave >> 10;
  const int s0 = st * 16;
  const int lane = threadIdx.x & 31;
  const float scale = 0.125f;  // 1/sqrt(HD)

  float Oself[32], Oknow[32];
  const size_t headS = (size_t)(b * NHD + h) * SEQ * HD;   // also v^T head offset
  const size_t headK = (size_t)(b * NHD + h) * KSEQ * HD;  // also kv^T head offset
  attn_branch(qb + headS, s0, kb + headS, vtb + headS, amask + b * SEQ, SEQ, scale,
              lds[wib], Oself);
  attn_branch(kqb + headS, s0, kkb + headK, kvtb + headK, emask + b * KSEQ, KSEQ,
              scale, lds[wib], Oknow);

  const int nlo = lane & 15;
  const int hi8 = (lane >> 4) << 3;
#pragma unroll
  for (int dt = 0; dt < 4; ++dt) {
#pragma unroll
    for (int j = 0; j < 8; ++j) {
      int s = s0 + j + hi8;
      int d = dt * 16 + nlo;
      out[(size_t)(b * SEQ + s) * HDIM + h * HD + d] =
          0.5f * (Oself[dt * 8 + j] + Oknow[dt * 8 + j]);
    }
  }
}

// ---------------- host launcher ----------------
extern "C" void kernel_launch(void* const* d_in, const int* in_sizes, int n_in,
                              void* d_out, int out_size, void* d_ws, size_t ws_size,
                              hipStream_t stream) {
  const float* hs    = (const float*)d_in[0];
  const float* ehs   = (const float*)d_in[1];
  const float* amask = (const float*)d_in[2];
  const float* emask = (const float*)d_in[3];
  const float* W[6];
  const float* bias[6];
  for (int i = 0; i < 6; ++i) {
    W[i]    = (const float*)d_in[4 + 2 * i];
    bias[i] = (const float*)d_in[5 + 2 * i];
  }
  float* out = (float*)d_out;

  // workspace carve-up (~64 MB total)
  char* ws = (char*)d_ws;
  size_t off = 0;
  auto alloc = [&](size_t bytes) -> char* {
    char* p = ws + off;
    off += (bytes + 255) & ~(size_t)255;
    return p;
  };
  bf16_t* hsb  = (bf16_t*)alloc((size_t)BSZ * SEQ * HDIM * 2);
  bf16_t* ehsb = (bf16_t*)alloc((size_t)BSZ * KSEQ * HDIM * 2);
  bf16_t* Wt[6];
  for (int i = 0; i < 6; ++i) Wt[i] = (bf16_t*)alloc((size_t)HDIM * HDIM * 2);
  bf16_t* qb   = (bf16_t*)alloc((size_t)BSZ * SEQ * HDIM * 2);
  bf16_t* kbuf = (bf16_t*)alloc((size_t)BSZ * SEQ * HDIM * 2);
  bf16_t* vtb  = (bf16_t*)alloc((size_t)BSZ * SEQ * HDIM * 2);
  bf16_t* kqb  = (bf16_t*)alloc((size_t)BSZ * SEQ * HDIM * 2);
  bf16_t* kkb  = (bf16_t*)alloc((size_t)BSZ * KSEQ * HDIM * 2);
  bf16_t* kvtb = (bf16_t*)alloc((size_t)BSZ * KSEQ * HDIM * 2);

  const int nh = BSZ * SEQ * HDIM;   // 4194304
  const int ne = BSZ * KSEQ * HDIM;  // 2097152
  cvt_f32_bf16<<<nh / 256, 256, 0, stream>>>(hs, hsb, nh);
  cvt_f32_bf16<<<ne / 256, 256, 0, stream>>>(ehs, ehsb, ne);
  for (int i = 0; i < 6; ++i)
    cvt_transpose_w<<<(HDIM * HDIM) / 256, 256, 0, stream>>>(W[i], Wt[i]);

  const int Mh = BSZ * SEQ;                        // 4096
  const int Me = BSZ * KSEQ;                       // 2048
  const int wavesH = (Mh / 32) * (HDIM / 64);      // 2048 waves
  const int wavesE = (Me / 32) * (HDIM / 64);      // 1024 waves
  gemm_proj<SEQ, 0><<<wavesH / 8, 256, 0, stream>>>(hsb,  Wt[0], bias[0], qb,   Mh);
  gemm_proj<SEQ, 0><<<wavesH / 8, 256, 0, stream>>>(hsb,  Wt[1], bias[1], kbuf, Mh);
  gemm_proj<SEQ, 1><<<wavesH / 8, 256, 0, stream>>>(hsb,  Wt[2], bias[2], vtb,  Mh);
  gemm_proj<SEQ, 0><<<wavesH / 8, 256, 0, stream>>>(hsb,  Wt[3], bias[3], kqb,  Mh);
  gemm_proj<KSEQ, 0><<<wavesE / 8, 256, 0, stream>>>(ehsb, Wt[4], bias[4], kkb,  Me);
  gemm_proj<KSEQ, 1><<<wavesE / 8, 256, 0, stream>>>(ehsb, Wt[5], bias[5], kvtb, Me);

  const int wavesA = BSZ * NHD * (SEQ / 16);  // 4096 waves, 8 per block
  attention_kernel<<<wavesA / 8, 256, 0, stream>>>(qb, kbuf, vtb, kqb, kkb, kvtb,
                                                   amask, emask, out);
}
